// GRU_45114336477964
// MI455X (gfx1250) — compile-verified, matrix operand order checked by reading, not run on hardware
//
#include <hip/hip_runtime.h>
#include <hip/hip_bf16.h>

typedef __attribute__((ext_vector_type(16))) __bf16 v16bf;
typedef __attribute__((ext_vector_type(8)))  float  v8f;

// ---------------------------------------------------------------------------
// Problem constants
// ---------------------------------------------------------------------------
#define GB_  64           // batch
#define GS_  1024         // seq len
#define GI_  512          // input size
#define GH_  512          // hidden size
#define GG_  1536         // 3*H

// ---------------------------------------------------------------------------
// bf16 WMMA fragment load from an LDS row (row-major, contiguous K).
// ISA 7.12.2, 16-bit A-matrix 16x32: lane l holds M=l&15; dword v holds the
// bf16 pair at K = kbase + ((v<4)?0:16) + ((l>=16)?8:0) + 2*(v&3).
// ---------------------------------------------------------------------------
union FragBF { unsigned u[8]; v16bf v; };

__device__ __forceinline__ v16bf load_frag_row(const __bf16* rowPtr, int kbase, int hi) {
  FragBF f;
#pragma unroll
  for (int v = 0; v < 8; ++v) {
    int k = kbase + ((v < 4) ? 0 : 16) + (hi ? 8 : 0) + ((v & 3) << 1);
    f.u[v] = *(const unsigned*)(rowPtr + k);   // 4-byte LDS load (k is even)
  }
  return f.v;
}

__device__ __forceinline__ v8f wmma_bf16(v16bf a, v16bf b, v8f c) {
  return __builtin_amdgcn_wmma_f32_16x16x32_bf16(false, a, false, b, (short)0, c,
                                                 false, false);
}

// ---------------------------------------------------------------------------
// fp32 -> bf16 conversion (grid-stride)
// ---------------------------------------------------------------------------
__global__ __launch_bounds__(256) void cvt_f32_bf16_kernel(const float* __restrict__ src,
                                                           __bf16* __restrict__ dst,
                                                           long n) {
  long i = (long)blockIdx.x * 256 + threadIdx.x;
  long stride = (long)gridDim.x * 256;
  for (; i < n; i += stride) dst[i] = (__bf16)src[i];
}

// ---------------------------------------------------------------------------
// Big input-to-hidden GEMM:
//   gx[m][g] = dot(Xrow(m), W[g][:]) + bias[g],  m = s*B + b, g in [0,1536)
// X is bf16. layer0_layout=1: Xrow(m) = x[b][s][:] at ((b*S)+s)*K.
//            layer0_layout=0: Xrow(m) at m*K (hs0 layout).
// Block: 256 thr, tile 64(M)x64(N), K-step 32, bf16 staged in LDS.
// ---------------------------------------------------------------------------
__global__ __launch_bounds__(256) void gemm_gx_kernel(const __bf16* __restrict__ X,
                                                      const __bf16* __restrict__ W,
                                                      const float* __restrict__ bias,
                                                      float* __restrict__ gx,
                                                      int K, int layer0_layout) {
  __shared__ __bf16 aLds[64][40];   // 32 K + 8 pad
  __shared__ __bf16 bLds[64][40];

  const int t    = threadIdx.x;
  const int mBase = blockIdx.y * 64;
  const int nBase = blockIdx.x * 64;
  const int lrow = t >> 2;          // 0..63
  const int lcol = (t & 3) * 8;     // 0,8,16,24 (bf16 elems; 16B chunks)

  long arow;
  {
    int m = mBase + lrow;
    if (layer0_layout) {
      int b = m & (GB_ - 1), s = m >> 6;          // m = s*B + b, B=64
      arow = ((long)b * GS_ + s) * (long)K;
    } else {
      arow = (long)m * K;
    }
  }
  const long brow = (long)(nBase + lrow) * K;

  const int wid  = t >> 5;
  const int lane = t & 31;
  const int wm   = wid & 3;         // 4 M sub-tiles of 16
  const int wn   = wid >> 2;        // 2 N sub-tiles of 32
  const int hi   = (lane >= 16);
  const int lr   = lane & 15;

  v8f acc0 = {}, acc1 = {};

  for (int k0 = 0; k0 < K; k0 += 32) {
    *(uint4*)&aLds[lrow][lcol] = *(const uint4*)(X + arow + k0 + lcol);
    *(uint4*)&bLds[lrow][lcol] = *(const uint4*)(W + brow + k0 + lcol);
    __syncthreads();
    v16bf a  = load_frag_row(&aLds[wm * 16 + lr][0], 0, hi);
    v16bf b0 = load_frag_row(&bLds[wn * 32 + lr][0], 0, hi);
    v16bf b1 = load_frag_row(&bLds[wn * 32 + 16 + lr][0], 0, hi);
    acc0 = wmma_bf16(a, b0, acc0);
    acc1 = wmma_bf16(a, b1, acc1);
    __syncthreads();
  }

  const int g0 = nBase + wn * 32 + lr;
  const int g1 = g0 + 16;
  const float bs0 = bias[g0];
  const float bs1 = bias[g1];
#pragma unroll
  for (int v = 0; v < 8; ++v) {
    int mg = mBase + wm * 16 + v + (hi ? 8 : 0);
    gx[(long)mg * GG_ + g0] = acc0[v] + bs0;
    gx[(long)mg * GG_ + g1] = acc1[v] + bs1;
  }
}

// ---------------------------------------------------------------------------
// Grid-wide barrier: sense-reversing atomic barrier (correct on plain launch)
// + CDNA5 cluster barrier (hardware-fast if dispatched as a 16-WG cluster,
// architectural NOP otherwise).
// ---------------------------------------------------------------------------
__device__ __forceinline__ void grid_barrier(unsigned* bar, unsigned nblk) {
  __syncthreads();
  __builtin_amdgcn_s_cluster_barrier();       // s_barrier_signal/wait -3 (NOP if not in cluster)
  if (threadIdx.x == 0) {
    __threadfence();
    volatile unsigned* gen = bar + 1;
    unsigned g = *gen;
    if (atomicAdd(bar, 1u) == nblk - 1) {
      bar[0] = 0u;
      __threadfence();
      atomicAdd((unsigned*)(bar + 1), 1u);
    } else {
      while (*gen == g) __builtin_amdgcn_s_sleep(2);
    }
    __threadfence();
  }
  __syncthreads();
}

// ---------------------------------------------------------------------------
// Persistent GRU scan. 16 blocks x 256 threads. Block bx owns hidden columns
// [32*bx, 32*bx+32). LDS (dynamic, 192000 B):
//   wlds : 96 rows x 520 bf16  (w_hh rows {j, j+512, j+1024 : j in slice})
//   hlds : 64 rows x 520 bf16  (per-step snapshot of h)
//   ghlds: 64 x 100 f32        (recurrent GEMM result)
// ---------------------------------------------------------------------------
#define WSTRIDE 520
#define SCAN_LDS_BYTES (96 * WSTRIDE * 2 + 64 * WSTRIDE * 2 + 64 * 100 * 4) // 192000

__global__ __launch_bounds__(256) void gru_scan_kernel(
    const float* __restrict__ gx,       // [S*B][1536], b_ih already folded in
    const __bf16* __restrict__ whh,     // [1536][512] bf16
    const float* __restrict__ bhh,      // [1536]
    float* __restrict__ hglob,          // [64][512] fp32 running state
    __bf16* __restrict__ hbf,           // [64][512] bf16 running state
    __bf16* __restrict__ hs_out,        // [S*B][512] bf16 or nullptr (layer0)
    float* __restrict__ out_f32,        // [B][S][512] or nullptr (layer1)
    float* __restrict__ hn_out,         // [64][512] final h for this layer
    unsigned* __restrict__ bar, int S) {
  extern __shared__ char smem[];
  __bf16* wlds  = (__bf16*)smem;                        // 96 x 520
  __bf16* hlds  = (__bf16*)(smem + 96 * WSTRIDE * 2);   // 64 x 520
  float*  ghlds = (float*)(smem + 160 * WSTRIDE * 2);   // 64 x 100

  const int t  = threadIdx.x;
  const int j0 = blockIdx.x * 32;

  // Preload the 96x512 bf16 weight slice into LDS (kept resident all scan).
  for (int i = t; i < 96 * 64; i += 256) {
    int row = i >> 6;
    int vc  = (i & 63) << 3;                       // 8 bf16 = 16B
    int grow = ((row >> 5) << 9) + j0 + (row & 31); // gate*512 + j0 + jj
    *(uint4*)(wlds + row * WSTRIDE + vc) = *(const uint4*)(whh + (long)grow * GH_ + vc);
  }
  __syncthreads();

  const int wid = t >> 5, lane = t & 31;
  const int wm = wid & 3;          // 4 M-tiles (batch rows)
  const int ng = wid >> 2;         // 2 groups of 3 N-tiles (6 tiles = 96 cols)
  const int hi = (lane >= 16), lr = lane & 15;

  for (int s = 0; s < S; ++s) {
    // Snapshot h (bf16) into LDS.
    for (int i = t; i < 64 * 64; i += 256) {
      int row = i >> 6, vc = (i & 63) << 3;
      *(uint4*)(hlds + row * WSTRIDE + vc) = *(const uint4*)(hbf + row * GH_ + vc);
    }
    grid_barrier(bar, gridDim.x);   // everyone has the old h before anyone writes

    // gh = h_prev(64x512) * W_slice^T(512x96), bf16 WMMA, f32 accum.
    v8f acc0 = {}, acc1 = {}, acc2 = {};
    const __bf16* arow = hlds + (wm * 16 + lr) * WSTRIDE;
    const __bf16* brow0 = wlds + ((ng * 3 + 0) * 16 + lr) * WSTRIDE;
    const __bf16* brow1 = wlds + ((ng * 3 + 1) * 16 + lr) * WSTRIDE;
    const __bf16* brow2 = wlds + ((ng * 3 + 2) * 16 + lr) * WSTRIDE;
    for (int k0 = 0; k0 < GH_; k0 += 32) {
      v16bf a = load_frag_row(arow, k0, hi);
      acc0 = wmma_bf16(a, load_frag_row(brow0, k0, hi), acc0);
      acc1 = wmma_bf16(a, load_frag_row(brow1, k0, hi), acc1);
      acc2 = wmma_bf16(a, load_frag_row(brow2, k0, hi), acc2);
    }
    {
      const int colbase = (ng * 3) * 16 + lr;
#pragma unroll
      for (int v = 0; v < 8; ++v) {
        int row = wm * 16 + v + (hi ? 8 : 0);
        ghlds[row * 100 + colbase]      = acc0[v];
        ghlds[row * 100 + colbase + 16] = acc1[v];
        ghlds[row * 100 + colbase + 32] = acc2[v];
      }
    }
    __syncthreads();

    // Gates + state update: 64x32 elements, 8 per thread.
    const float* gxs = gx + (long)s * GB_ * GG_;
#pragma unroll
    for (int e = 0; e < 8; ++e) {
      int idx = t * 8 + e;
      int b = idx >> 5, jj = idx & 31;
      const float* gb = gxs + b * GG_;
      float xr = gb[j0 + jj];
      float xz = gb[GH_ + j0 + jj];
      float xn = gb[2 * GH_ + j0 + jj];
      float ghr = ghlds[b * 100 + jj];
      float ghz = ghlds[b * 100 + 32 + jj];
      float ghn = ghlds[b * 100 + 64 + jj];
      float r = 1.0f / (1.0f + __expf(-(xr + ghr + bhh[j0 + jj])));
      float z = 1.0f / (1.0f + __expf(-(xz + ghz + bhh[GH_ + j0 + jj])));
      float n = tanhf(xn + r * (ghn + bhh[2 * GH_ + j0 + jj]));
      float hp = hglob[b * GH_ + j0 + jj];
      float hv = (1.0f - z) * n + z * hp;
      hglob[b * GH_ + j0 + jj] = hv;
      hbf[b * GH_ + j0 + jj] = (__bf16)hv;
      if (hs_out)  hs_out[((long)(s * GB_ + b)) * GH_ + j0 + jj] = (__bf16)hv;
      if (out_f32) out_f32[((long)b * S + s) * GH_ + j0 + jj] = hv;
    }
    grid_barrier(bar, gridDim.x);   // h writes visible before next snapshot
  }

  // Final hidden state for h_n.
#pragma unroll
  for (int e = 0; e < 8; ++e) {
    int idx = t * 8 + e;
    int b = idx >> 5, jj = idx & 31;
    hn_out[b * GH_ + j0 + jj] = hglob[b * GH_ + j0 + jj];
  }
}

// ---------------------------------------------------------------------------
// Zero the running state + barrier words before each layer scan.
// ---------------------------------------------------------------------------
__global__ __launch_bounds__(256) void init_state_kernel(float* hglob, __bf16* hbf,
                                                         unsigned* bar) {
  int i = blockIdx.x * 256 + threadIdx.x;
  if (i < GB_ * GH_) {
    hglob[i] = 0.0f;
    hbf[i] = (__bf16)0.0f;
  }
  if (i < 8) bar[i] = 0u;
}

// ---------------------------------------------------------------------------
// Host-side orchestration
// ---------------------------------------------------------------------------
extern "C" void kernel_launch(void* const* d_in, const int* in_sizes, int n_in,
                              void* d_out, int out_size, void* d_ws, size_t ws_size,
                              hipStream_t stream) {
  (void)in_sizes; (void)n_in; (void)out_size; (void)ws_size;

  const float* x     = (const float*)d_in[0];
  const float* w_ih0 = (const float*)d_in[1];
  const float* w_hh0 = (const float*)d_in[2];
  const float* b_ih0 = (const float*)d_in[3];
  const float* b_hh0 = (const float*)d_in[4];
  const float* w_ih1 = (const float*)d_in[5];
  const float* w_hh1 = (const float*)d_in[6];
  const float* b_ih1 = (const float*)d_in[7];
  const float* b_hh1 = (const float*)d_in[8];
  float* out = (float*)d_out;   // [B*S*H] output, then [2*B*H] h_n

  // Workspace carve-up.
  char* ws = (char*)d_ws;
  size_t off = 0;
  auto take = [&](size_t bytes) -> char* {
    char* p = ws + off;
    off += (bytes + 255) & ~(size_t)255;
    return p;
  };
  const long nX   = (long)GB_ * GS_ * GI_;          // 33,554,432
  const long nWih = (long)GG_ * GI_;                // 786,432
  const long nWhh = (long)GG_ * GH_;
  __bf16*   xbf   = (__bf16*)take((size_t)nX * 2);
  __bf16*   wih0b = (__bf16*)take((size_t)nWih * 2);
  __bf16*   whh0b = (__bf16*)take((size_t)nWhh * 2);
  __bf16*   wih1b = (__bf16*)take((size_t)nWhh * 2);
  __bf16*   whh1b = (__bf16*)take((size_t)nWhh * 2);
  float*    gx    = (float*)take((size_t)GS_ * GB_ * GG_ * 4);  // 402 MB, reused
  __bf16*   hs0b  = (__bf16*)take((size_t)GS_ * GB_ * GH_ * 2);
  float*    hglob = (float*)take((size_t)GB_ * GH_ * 4);
  __bf16*   hbf   = (__bf16*)take((size_t)GB_ * GH_ * 2);
  unsigned* bar   = (unsigned*)take(256);

  float* out_hn = out + (long)GB_ * GS_ * GH_;

  // 1) fp32 -> bf16 conversions.
  cvt_f32_bf16_kernel<<<4096, 256, 0, stream>>>(x, xbf, nX);
  cvt_f32_bf16_kernel<<<768, 256, 0, stream>>>(w_ih0, wih0b, nWih);
  cvt_f32_bf16_kernel<<<768, 256, 0, stream>>>(w_hh0, whh0b, nWhh);
  cvt_f32_bf16_kernel<<<768, 256, 0, stream>>>(w_ih1, wih1b, nWhh);
  cvt_f32_bf16_kernel<<<768, 256, 0, stream>>>(w_hh1, whh1b, nWhh);

  // Allow >64KB dynamic LDS for the persistent scan kernel.
  (void)hipFuncSetAttribute((const void*)gru_scan_kernel,
                            hipFuncAttributeMaxDynamicSharedMemorySize,
                            SCAN_LDS_BYTES);

  dim3 gemm_grid(GG_ / 64, (GS_ * GB_) / 64);   // (24, 1024)

  // 2) Layer 0: gx = x * w_ih0^T + b_ih0 ; then recurrent scan.
  gemm_gx_kernel<<<gemm_grid, 256, 0, stream>>>(xbf, wih0b, b_ih0, gx, GI_, 1);
  init_state_kernel<<<128, 256, 0, stream>>>(hglob, hbf, bar);
  gru_scan_kernel<<<16, 256, SCAN_LDS_BYTES, stream>>>(
      gx, whh0b, b_hh0, hglob, hbf, hs0b, nullptr, out_hn, bar, GS_);

  // 3) Layer 1: gx = hs0 * w_ih1^T + b_ih1 ; scan writes output + h_n[1].
  gemm_gx_kernel<<<gemm_grid, 256, 0, stream>>>(hs0b, wih1b, b_ih1, gx, GH_, 0);
  init_state_kernel<<<128, 256, 0, stream>>>(hglob, hbf, bar);
  gru_scan_kernel<<<16, 256, SCAN_LDS_BYTES, stream>>>(
      gx, whh1b, b_hh1, hglob, hbf, nullptr, out, out_hn + (long)GB_ * GH_, bar, GS_);
}
